// Model_25031069401684
// MI455X (gfx1250) — compile-verified, matrix operand order checked by reading, not run on hardware
//
#include <hip/hip_runtime.h>
#include <hip/hip_bf16.h>

typedef __attribute__((ext_vector_type(16))) _Float16 v16h;
typedef __attribute__((ext_vector_type(8)))  float    v8f;

#define HID 64
#define NODE_D 32

// ---------------------------------------------------------------------------
// Degree (scatter count) and inverse-degree
// ---------------------------------------------------------------------------
__global__ void deg_kernel(const long long* __restrict__ ei, int n_edges,
                           float* __restrict__ deg) {
    int i = blockIdx.x * blockDim.x + threadIdx.x;
    if (i < n_edges) {
        int d = (int)ei[(size_t)n_edges + i];   // row 1 = dst
        atomicAdd(&deg[d], 1.0f);
    }
}

__global__ void invdeg_kernel(float* __restrict__ deg, int n_nodes) {
    int i = blockIdx.x * blockDim.x + threadIdx.x;
    if (i < n_nodes) {
        float d = deg[i];
        deg[i] = d > 0.f ? 1.f / d : 0.f;
    }
}

// ---------------------------------------------------------------------------
// h = relu(x @ lin_w.T + lin_b)   via v_wmma_f32_16x16x32_f16
// One wave computes a 16-row x 64-col tile = 4 WMMA ops (K=32 in one shot).
// m0 is clamped so the whole tile is in-range (requires n_nodes >= 16);
// overlapping tail waves rewrite identical values -> stores are UNGUARDED
// straight-line code (no exec-mask churn in the epilogue).
// A (16x32 f16): lane L holds row M=L%16; half t -> K = (t/8)*16 + (L>=16)*8 + t%8
// B (32x16 f16): symmetric with N in place of M; B[k][n] = lin_w[n*32+k]
// C/D (16x16 f32): vgpr r, lane L -> row M = r + 8*(L>=16), col N = L%16
// ---------------------------------------------------------------------------
__global__ void lin_relu_wmma(const float* __restrict__ x,
                              const float* __restrict__ w,     // [64,32]
                              const float* __restrict__ bias,  // [64]
                              float* __restrict__ h, int n_nodes) {
    const int lane = threadIdx.x & 31;
    const int wave = blockIdx.x * (blockDim.x >> 5) + (threadIdx.x >> 5);

    int m0 = wave * 16;
    if (m0 + 16 > n_nodes) m0 = (n_nodes > 16) ? (n_nodes - 16) : 0;

    const int mr = lane & 15;
    const int hi = lane >> 4;       // 0 or 1
    const int kb = hi * 8;

    // A fragment: row m0+mr, two contiguous 8-float runs -> b128-vectorizable
    const float* xr = x + (size_t)(m0 + mr) * NODE_D;
    v16h a;
#pragma unroll
    for (int t = 0; t < 16; ++t) {
        int K = ((t >> 3) << 4) + kb + (t & 7);
        a[t] = (_Float16)xr[K];
    }

    // base address of this lane's output column for row m0 + hi*8
    float* hp = h + (size_t)(m0 + hi * 8) * HID + mr;

#pragma unroll
    for (int nt = 0; nt < 4; ++nt) {
        const int n = nt * 16 + mr;
        const float* wr = w + (size_t)n * NODE_D;
        v16h b;
#pragma unroll
        for (int t = 0; t < 16; ++t) {
            int K = ((t >> 3) << 4) + kb + (t & 7);
            b[t] = (_Float16)wr[K];
        }
        v8f c = {};
        c = __builtin_amdgcn_wmma_f32_16x16x32_f16(
                /*neg_a=*/false, a, /*neg_b=*/false, b,
                /*c_mod=*/(short)0, c, /*reuse_a=*/false, /*reuse_b=*/false);

        const float bv = bias[n];
#pragma unroll
        for (int r = 0; r < 8; ++r) {
            float v = c[r] + bv;
            hp[(size_t)r * HID + nt * 16] = v > 0.f ? v : 0.f;
        }
    }
}

// ---------------------------------------------------------------------------
// Scatter-add: msg[dst] += h[src]   (16 threads/edge, float4 per thread)
// h and msg are 25.6 MB each -> resident in the 192 MB L2; atomics hit L2.
// ---------------------------------------------------------------------------
__global__ void scatter_kernel(const long long* __restrict__ ei, int n_edges,
                               const float* __restrict__ h,
                               float* __restrict__ msg) {
    long long idx = (long long)blockIdx.x * blockDim.x + threadIdx.x;
    long long total = (long long)n_edges * 16;
    if (idx >= total) return;
    int e = (int)(idx >> 4);
    int q = (int)(idx & 15);
    int s = (int)ei[e];
    int d = (int)ei[(size_t)n_edges + e];
    const float4 v = *(const float4*)(h + (size_t)s * HID + q * 4);
    float* mp = msg + (size_t)d * HID + q * 4;
    atomicAdd(mp + 0, v.x);
    atomicAdd(mp + 1, v.y);
    atomicAdd(mp + 2, v.z);
    atomicAdd(mp + 3, v.w);
}

// ---------------------------------------------------------------------------
// h = (h + msg * inv_deg) * 0.5   (float4 per thread)
// ---------------------------------------------------------------------------
__global__ void update_kernel(float* __restrict__ h, const float* __restrict__ msg,
                              const float* __restrict__ inv_deg, int n_nodes) {
    int idx = blockIdx.x * blockDim.x + threadIdx.x;
    int total = n_nodes * (HID / 4);
    if (idx >= total) return;
    int n = idx >> 4;
    float inv = inv_deg[n];
    float4 hv = ((const float4*)h)[idx];
    float4 mv = ((const float4*)msg)[idx];
    hv.x = (hv.x + mv.x * inv) * 0.5f;
    hv.y = (hv.y + mv.y * inv) * 0.5f;
    hv.z = (hv.z + mv.z * inv) * 0.5f;
    hv.w = (hv.w + mv.w * inv) * 0.5f;
    ((float4*)h)[idx] = hv;
}

// ---------------------------------------------------------------------------
// accum[c] += sum_rows relu(h[r][c])  (LDS-reduced per block, 64 atomics/block)
// ---------------------------------------------------------------------------
__global__ void colsum_kernel(const float* __restrict__ h, int n_nodes,
                              float* __restrict__ accum) {
    __shared__ float lds[HID];
    if (threadIdx.x < HID) lds[threadIdx.x] = 0.f;
    __syncthreads();
    const int c = threadIdx.x & (HID - 1);
    const int rpb = blockDim.x / HID;  // 4 rows per block pass
    const int stride = gridDim.x * rpb;
    float s = 0.f;
    for (int r = blockIdx.x * rpb + (threadIdx.x >> 6); r < n_nodes; r += stride) {
        float v = h[(size_t)r * HID + c];
        s += v > 0.f ? v : 0.f;
    }
    atomicAdd(&lds[c], s);
    __syncthreads();
    if (threadIdx.x < HID) atomicAdd(&accum[threadIdx.x], lds[threadIdx.x]);
}

// ---------------------------------------------------------------------------
// out = (accum/N) @ pool_w.T @ read_w.T + read_b    (1 wave; shfl reduction)
// ---------------------------------------------------------------------------
__global__ void readout_kernel(const float* __restrict__ accum,
                               const float* __restrict__ pool_w,  // [32,64]
                               const float* __restrict__ read_w,  // [1,32]
                               const float* __restrict__ read_b,  // [1]
                               int n_nodes, float* __restrict__ out) {
    const int j = threadIdx.x;  // 0..31
    float g = 0.f;
#pragma unroll
    for (int k = 0; k < HID; ++k) g += accum[k] * pool_w[j * HID + k];
    g *= (1.0f / (float)n_nodes);
    float p = g * read_w[j];
#pragma unroll
    for (int off = 16; off > 0; off >>= 1) p += __shfl_down(p, off, 32);
    if (j == 0) out[0] = p + read_b[0];
}

// ---------------------------------------------------------------------------
extern "C" void kernel_launch(void* const* d_in, const int* in_sizes, int n_in,
                              void* d_out, int out_size, void* d_ws, size_t ws_size,
                              hipStream_t stream) {
    const float*     x      = (const float*)d_in[0];
    const long long* ei     = (const long long*)d_in[1];   // int64 [2, E]
    const float*     lin_w  = (const float*)d_in[2];
    const float*     lin_b  = (const float*)d_in[3];
    const float*     pool_w = (const float*)d_in[4];
    const float*     read_w = (const float*)d_in[5];
    const float*     read_b = (const float*)d_in[6];
    float*           out    = (float*)d_out;

    const int n_nodes = in_sizes[0] / NODE_D;
    const int n_edges = in_sizes[1] / 2;

    // workspace layout (floats): h | msg | inv_deg | accum
    float* h       = (float*)d_ws;
    float* msg     = h + (size_t)n_nodes * HID;
    float* inv_deg = msg + (size_t)n_nodes * HID;
    float* accum   = inv_deg + (((size_t)n_nodes + 3) & ~(size_t)3);

    // 1) degrees -> inv_deg
    hipMemsetAsync(inv_deg, 0, (size_t)n_nodes * sizeof(float), stream);
    deg_kernel<<<(n_edges + 255) / 256, 256, 0, stream>>>(ei, n_edges, inv_deg);
    invdeg_kernel<<<(n_nodes + 255) / 256, 256, 0, stream>>>(inv_deg, n_nodes);

    // 2) h = relu(x @ lin_w.T + lin_b) via WMMA (8 waves/block, 16 rows/wave)
    const int tiles = (n_nodes + 15) / 16;
    lin_relu_wmma<<<(tiles + 7) / 8, 256, 0, stream>>>(x, lin_w, lin_b, h, n_nodes);

    // 3) 4 message-passing steps
    const long long sthreads = (long long)n_edges * 16;
    const int sblocks = (int)((sthreads + 255) / 256);
    const int ublocks = (n_nodes * (HID / 4) + 255) / 256;
    for (int step = 0; step < 4; ++step) {
        hipMemsetAsync(msg, 0, (size_t)n_nodes * HID * sizeof(float), stream);
        scatter_kernel<<<sblocks, 256, 0, stream>>>(ei, n_edges, h, msg);
        update_kernel<<<ublocks, 256, 0, stream>>>(h, msg, inv_deg, n_nodes);
    }

    // 4) pooled readout: mean commutes with the linear maps -> 64-wide colsum
    hipMemsetAsync(accum, 0, HID * sizeof(float), stream);
    colsum_kernel<<<2048, 256, 0, stream>>>(h, n_nodes, accum);
    readout_kernel<<<1, 32, 0, stream>>>(accum, pool_w, read_w, read_b, n_nodes, out);
}
